// StandardAttention_32134945309253
// MI455X (gfx1250) — matrix operand order repaired and checked by reading, not compiled, not run on hardware
//
#include <hip/hip_runtime.h>
#include <stdint.h>

// ---------------------------------------------------------------------------
// Problem constants (from the reference): B=4, T=2048, C=1024, H=16, Dh=64
// ---------------------------------------------------------------------------
#define BB 4
#define TT 2048
#define CC 1024
#define HH 16
#define DH 64
#define NROWS (BB * TT)          // 8192 token rows
#define QK_SCALE 0.125f          // 1/sqrt(64), folded into Q at QKV epilogue

typedef __attribute__((ext_vector_type(16))) __bf16 bf16x16;
typedef __attribute__((ext_vector_type(8)))  float  f32x8;

union Frag16 { bf16x16 v; uint32_t u[8]; };

__device__ __forceinline__ f32x8 wmma_bf16(const bf16x16 a, const bf16x16 b, const f32x8 c) {
  // (neg_a, A, neg_b, B, c_mod, C, reuse_a, reuse_b)
  return __builtin_amdgcn_wmma_f32_16x16x32_bf16(false, a, false, b, (short)0, c, false, false);
}

__device__ __forceinline__ f32x8 zero8() {
  f32x8 z;
#pragma unroll
  for (int i = 0; i < 8; ++i) z[i] = 0.0f;
  return z;
}

__device__ __forceinline__ uint16_t f2bf(float x) {   // round-to-nearest-even
  uint32_t u = __float_as_uint(x);
  return (uint16_t)((u + 0x7FFFu + ((u >> 16) & 1u)) >> 16);
}

// reductions across the 16 lanes of a half-wave (masks 1,2,4,8 stay in-half)
__device__ __forceinline__ float rowmax16(float x) {
#pragma unroll
  for (int m = 8; m >= 1; m >>= 1) x = fmaxf(x, __shfl_xor(x, m, 32));
  return x;
}
__device__ __forceinline__ float rowsum16(float x) {
#pragma unroll
  for (int m = 8; m >= 1; m >>= 1) x += __shfl_xor(x, m, 32);
  return x;
}

// ---------------------------------------------------------------------------
// fp32 -> bf16 conversion
// ---------------------------------------------------------------------------
__global__ void cvt_f32_bf16(const float* __restrict__ src, uint16_t* __restrict__ dst, int n) {
  int i = blockIdx.x * blockDim.x + threadIdx.x;
  if (i < n) dst[i] = f2bf(src[i]);
}

// ---------------------------------------------------------------------------
// One wave computes a 32(M) x 64(N) tile of  A[M,K] * W[N,K]^T, K multiple of 32.
// A row-major bf16 (lda = K), W row-major [N,K] bf16 (so B-columns are W rows).
// acc[mi*4+ni] holds the 16x16 tile at (tileM+16*mi, tileN+16*ni).
// A-frag (16-bit 16x32): lane (half,r): row = r, dword v -> k-dword v+8*half
// B-frag (16-bit 32x16): lane (half,r): col = r, dword v -> k rows 16*half+2v,+1
// ---------------------------------------------------------------------------
__device__ __forceinline__ void gemm_wave_32x64(const uint16_t* __restrict__ A,
                                                const uint16_t* __restrict__ W,
                                                int K, int tileM, int tileN,
                                                int lane, f32x8 acc[8]) {
  const int half = lane >> 4, r = lane & 15;
#pragma unroll
  for (int i = 0; i < 8; ++i) acc[i] = zero8();

  for (int k0 = 0; k0 < K; k0 += 32) {
    Frag16 a0, a1;
    const uint32_t* Ar0 = (const uint32_t*)(A + (size_t)(tileM + r) * K + k0);
    const uint32_t* Ar1 = (const uint32_t*)(A + (size_t)(tileM + 16 + r) * K + k0);
#pragma unroll
    for (int v = 0; v < 8; ++v) {
      const int di = v + 8 * half;
      a0.u[v] = Ar0[di];
      a1.u[v] = Ar1[di];
    }
#pragma unroll
    for (int ni = 0; ni < 4; ++ni) {
      Frag16 bf;
      const uint32_t* Wr = (const uint32_t*)(W + (size_t)(tileN + ni * 16 + r) * K + k0);
#pragma unroll
      for (int v = 0; v < 8; ++v) bf.u[v] = Wr[8 * half + v];
      acc[ni]     = wmma_bf16(a0.v, bf.v, acc[ni]);
      acc[4 + ni] = wmma_bf16(a1.v, bf.v, acc[4 + ni]);
    }
  }
}

// ---------------------------------------------------------------------------
// Stage 1: qkv = x @ W_qkv^T, scattered to:
//   Q [b,h,t,d]  (pre-scaled by 1/sqrt(Dh))
//   K [b,h,t,d]
//   V [b,h,d,t]  (TRANSPOSED: so the P*V B-fragments are contiguous loads)
// grid = (NROWS/64, 3072/256), block = 256 (8 waves = 2M x 4N)
// ---------------------------------------------------------------------------
__global__ __launch_bounds__(256) void qkv_gemm_kernel(const uint16_t* __restrict__ xb,
                                                       const uint16_t* __restrict__ Wqb,
                                                       uint16_t* __restrict__ Qb,
                                                       uint16_t* __restrict__ Kb,
                                                       uint16_t* __restrict__ Vtb) {
  const int wave = threadIdx.x >> 5, lane = threadIdx.x & 31;
  const int mw = wave >> 2, nw = wave & 3;
  const int tileM = blockIdx.x * 64 + mw * 32;
  const int tileN = blockIdx.y * 256 + nw * 64;

  f32x8 acc[8];
  gemm_wave_32x64(xb, Wqb, CC, tileM, tileN, lane, acc);

  const int half = lane >> 4, r = lane & 15;
#pragma unroll
  for (int ni = 0; ni < 4; ++ni) {
    const int o   = tileN + ni * 16 + r;        // output feature in [0,3072)
    const int sel = o >> 10;                    // 0=Q 1=K 2=V (wave-uniform per ni)
    const int hd  = o & 1023;
    const int h   = hd >> 6, d = hd & 63;
    const float scale = (sel == 0) ? QK_SCALE : 1.0f;
#pragma unroll
    for (int mi = 0; mi < 2; ++mi) {
#pragma unroll
      for (int v = 0; v < 8; ++v) {
        const int i = tileM + mi * 16 + v + 8 * half;   // token row
        const int b = i >> 11, t = i & 2047;
        const uint16_t val = f2bf(acc[mi * 4 + ni][v] * scale);
        const size_t bh = (size_t)(b * HH + h);
        if (sel == 0)      Qb [(bh * TT + t) * DH + d] = val;
        else if (sel == 1) Kb [(bh * TT + t) * DH + d] = val;
        else               Vtb[(bh * DH + d) * TT + t] = val;   // transposed
      }
    }
  }
}

// ---------------------------------------------------------------------------
// Stage 2: flash-style causal attention, one wave per 16 query rows.
// grid = (TT/128, BB*HH), block = 256 (8 waves). Output Ob is [b,t,h*64+d] bf16.
// V is read from the transposed layout [b,h,d,t]: each B-fragment column
// (fixed d, consecutive keys) is contiguous -> vectorizes to global_load_b128.
// ---------------------------------------------------------------------------
__global__ __launch_bounds__(256) void attn_kernel(const uint16_t* __restrict__ Qb,
                                                   const uint16_t* __restrict__ Kb,
                                                   const uint16_t* __restrict__ Vtb,
                                                   uint16_t* __restrict__ Ob) {
  __shared__ uint16_t Plds[8][16 * 32];         // per-wave 16x32 bf16 P tile
  const int wave = threadIdx.x >> 5, lane = threadIdx.x & 31;
  const int half = lane >> 4, r = lane & 15;
  const int bh = blockIdx.y;
  const int b = bh >> 4, h = bh & 15;
  const int qw = blockIdx.x * 128 + wave * 16;  // first query row of this wave

  const uint16_t* Qp  = Qb  + (size_t)bh * TT * DH;
  const uint16_t* Kp  = Kb  + (size_t)bh * TT * DH;
  const uint16_t* Vtp = Vtb + (size_t)bh * DH * TT;

  // Q A-fragments for d=[0,32) and d=[32,64)
  Frag16 qa0, qa1;
  {
    const uint32_t* Qrow = (const uint32_t*)(Qp + (size_t)(qw + r) * DH);
#pragma unroll
    for (int v = 0; v < 8; ++v) {
      const int di = v + 8 * half;
      qa0.u[v] = Qrow[di];
      qa1.u[v] = Qrow[16 + di];
    }
  }

  f32x8 oac[4];
#pragma unroll
  for (int n = 0; n < 4; ++n) oac[n] = zero8();
  float mrow[8], lrow[8];                       // per C-row (row = v + 8*half)
#pragma unroll
  for (int v = 0; v < 8; ++v) { mrow[v] = -3.0e38f; lrow[v] = 0.0f; }

  const int ktiles = (qw + 15) / 32 + 1;        // causal: keys up to qw+15

  for (int kt = 0; kt < ktiles; ++kt) {
    const int kbase = kt * 32;

    // ---- S = Q K^T for 32 keys (two 16-col C tiles) ----
    f32x8 s0 = zero8(), s1 = zero8();
    {
      const uint32_t* Kr0 = (const uint32_t*)(Kp + (size_t)(kbase + r) * DH);
      const uint32_t* Kr1 = (const uint32_t*)(Kp + (size_t)(kbase + 16 + r) * DH);
      Frag16 kb00, kb01, kb10, kb11;
#pragma unroll
      for (int v = 0; v < 8; ++v) {
        kb00.u[v] = Kr0[8 * half + v];      kb01.u[v] = Kr0[16 + 8 * half + v];
        kb10.u[v] = Kr1[8 * half + v];      kb11.u[v] = Kr1[16 + 8 * half + v];
      }
      s0 = wmma_bf16(qa0.v, kb00.v, s0);  s0 = wmma_bf16(qa1.v, kb01.v, s0);
      s1 = wmma_bf16(qa0.v, kb10.v, s1);  s1 = wmma_bf16(qa1.v, kb11.v, s1);
    }

    // ---- causal mask + online softmax ----
#pragma unroll
    for (int v = 0; v < 8; ++v) {
      const int q  = qw + v + 8 * half;
      const int k0i = kbase + r, k1i = kbase + 16 + r;
      float a0 = (k0i <= q) ? s0[v] : -3.0e38f;
      float a1 = (k1i <= q) ? s1[v] : -3.0e38f;
      float mt = rowmax16(fmaxf(a0, a1));
      const float mn = fmaxf(mrow[v], mt);
      const float alpha = __expf(mrow[v] - mn);
      mrow[v] = mn;
      const float p0 = __expf(a0 - mn);
      const float p1 = __expf(a1 - mn);
      s0[v] = p0; s1[v] = p1;
      lrow[v] = lrow[v] * alpha + rowsum16(p0 + p1);
#pragma unroll
      for (int n = 0; n < 4; ++n) oac[n][v] *= alpha;
    }

    // ---- P (C layout) -> LDS -> A-fragment layout ----
    uint16_t* pl = &Plds[wave][0];
#pragma unroll
    for (int v = 0; v < 8; ++v) {
      const int row = v + 8 * half;
      pl[row * 32 + r]      = f2bf(s0[v]);
      pl[row * 32 + 16 + r] = f2bf(s1[v]);
    }
    Frag16 pf;                                  // same-wave LDS RAW: dscnt-ordered
    {
      const uint32_t* Pr = (const uint32_t*)&Plds[wave][r * 32];
#pragma unroll
      for (int v = 0; v < 8; ++v) pf.u[v] = Pr[v + 8 * half];
    }

    // ---- O += P V (4 N-tiles over Dh); V transposed -> contiguous B loads ----
#pragma unroll
    for (int ni = 0; ni < 4; ++ni) {
      Frag16 vf;
      const int d = ni * 16 + r;
      const uint32_t* Vr = (const uint32_t*)(Vtp + (size_t)d * TT + kbase);
#pragma unroll
      for (int v = 0; v < 8; ++v) vf.u[v] = Vr[8 * half + v];
      oac[ni] = wmma_bf16(pf.v, vf.v, oac[ni]);
    }
  }

  // ---- normalize and store O as [b, t, h*64+d] bf16 ----
#pragma unroll
  for (int v = 0; v < 8; ++v) {
    const int t = qw + v + 8 * half;
    const float inv = 1.0f / lrow[v];
#pragma unroll
    for (int ni = 0; ni < 4; ++ni) {
      Ob[((size_t)(b * TT + t)) * CC + h * DH + ni * 16 + r] = f2bf(oac[ni][v] * inv);
    }
  }
}

// ---------------------------------------------------------------------------
// Stage 3: out = O @ W_o^T, fp32 result straight to d_out.
// grid = (NROWS/64, CC/256), block = 256
// ---------------------------------------------------------------------------
__global__ __launch_bounds__(256) void oproj_gemm_kernel(const uint16_t* __restrict__ Ob,
                                                         const uint16_t* __restrict__ Wob,
                                                         float* __restrict__ out) {
  const int wave = threadIdx.x >> 5, lane = threadIdx.x & 31;
  const int mw = wave >> 2, nw = wave & 3;
  const int tileM = blockIdx.x * 64 + mw * 32;
  const int tileN = blockIdx.y * 256 + nw * 64;

  f32x8 acc[8];
  gemm_wave_32x64(Ob, Wob, CC, tileM, tileN, lane, acc);

  const int half = lane >> 4, r = lane & 15;
#pragma unroll
  for (int mi = 0; mi < 2; ++mi) {
#pragma unroll
    for (int ni = 0; ni < 4; ++ni) {
#pragma unroll
      for (int v = 0; v < 8; ++v) {
        const int i = tileM + mi * 16 + v + 8 * half;
        out[(size_t)i * CC + tileN + ni * 16 + r] = acc[mi * 4 + ni][v];
      }
    }
  }
}

// ---------------------------------------------------------------------------
// Host side
// ---------------------------------------------------------------------------
extern "C" void kernel_launch(void* const* d_in, const int* in_sizes, int n_in,
                              void* d_out, int out_size, void* d_ws, size_t ws_size,
                              hipStream_t stream) {
  (void)in_sizes; (void)n_in; (void)out_size; (void)ws_size;
  const float* x    = (const float*)d_in[0];   // [4,2048,1024]
  const float* Wqkv = (const float*)d_in[1];   // [3072,1024]
  const float* Wo   = (const float*)d_in[2];   // [1024,1024]
  float* out = (float*)d_out;                  // [4,2048,1024]

  uint8_t* ws = (uint8_t*)d_ws;
  size_t off = 0;
  uint16_t* xb  = (uint16_t*)(ws + off); off += (size_t)NROWS * CC * 2;          // 16 MB
  uint16_t* Wqb = (uint16_t*)(ws + off); off += (size_t)3 * CC * CC * 2;         //  6 MB
  uint16_t* Wob = (uint16_t*)(ws + off); off += (size_t)CC * CC * 2;             //  2 MB
  uint16_t* Qb  = (uint16_t*)(ws + off); off += (size_t)BB * HH * TT * DH * 2;   // 16 MB
  uint16_t* Kb  = (uint16_t*)(ws + off); off += (size_t)BB * HH * TT * DH * 2;   // 16 MB
  uint16_t* Vtb = (uint16_t*)(ws + off); off += (size_t)BB * HH * TT * DH * 2;   // 16 MB (transposed)
  uint16_t* Ob  = (uint16_t*)(ws + off); off += (size_t)NROWS * CC * 2;          // 16 MB

  const int nx = NROWS * CC, nwq = 3 * CC * CC, nwo = CC * CC;
  cvt_f32_bf16<<<(nx  + 255) / 256, 256, 0, stream>>>(x,    xb,  nx);
  cvt_f32_bf16<<<(nwq + 255) / 256, 256, 0, stream>>>(Wqkv, Wqb, nwq);
  cvt_f32_bf16<<<(nwo + 255) / 256, 256, 0, stream>>>(Wo,   Wob, nwo);

  qkv_gemm_kernel<<<dim3(NROWS / 64, (3 * CC) / 256), 256, 0, stream>>>(xb, Wqb, Qb, Kb, Vtb);
  attn_kernel<<<dim3(TT / 128, BB * HH), 256, 0, stream>>>(Qb, Kb, Vtb, Ob);
  oproj_gemm_kernel<<<dim3(NROWS / 64, CC / 256), 256, 0, stream>>>(Ob, Wob, out);
}
